// NodeModel_91122026152383
// MI455X (gfx1250) — compile-verified, matrix operand order checked by reading, not run on hardware
//
#include <hip/hip_runtime.h>
#include <hip/hip_bf16.h>

// ---------------------------------------------------------------------------
// MetaLayer GNN step for MI455X (gfx1250, wave32, WMMA bf16 w/ f32 accum)
// M=32 register blocking per wave, N-tile loop unrolled x2:
// each kc iteration = 2 B-fragment loads + 4 WMMAs (deep load/compute overlap)
// ---------------------------------------------------------------------------

#define NN 50000      // nodes
#define NE 800000     // edges
#define DN 128        // node feature dim
#define DE 128        // edge feature dim
#define DH 256        // hidden
#define DO 256        // output

typedef __bf16 bf16;
typedef __attribute__((ext_vector_type(16))) __bf16 v16bf;
typedef __attribute__((ext_vector_type(8)))  __bf16 v8bf;
typedef __attribute__((ext_vector_type(4)))  __bf16 v4bf;
typedef __attribute__((ext_vector_type(8)))  float  v8f;

#define ZERO8 {0.f,0.f,0.f,0.f,0.f,0.f,0.f,0.f}
#define WMMA_BF16(A, B, C) \
    __builtin_amdgcn_wmma_f32_16x16x32_bf16(false, (A), false, (B), (short)0, (C), false, false)

// ---- A-fragment loader (bf16 16x32 layout, ISA 7.12.2) --------------------
// lane l holds row M = l&15 ; K base gets +8 for lanes 16..31.
// elements 0..7  -> K = k0 .. k0+7 ; elements 8..15 -> K = k0+16 .. k0+23
__device__ __forceinline__ v16bf load_a_frag(const bf16* __restrict__ row, int k0) {
    v8bf lo = *(const v8bf*)(row + k0);
    v8bf hi = *(const v8bf*)(row + k0 + 16);
    return __builtin_shufflevector(lo, hi, 0,1,2,3,4,5,6,7,8,9,10,11,12,13,14,15);
}

__device__ __forceinline__ float elu1(float t) {
    return t > 0.f ? t : (__expf(t) - 1.f);
}

__device__ __forceinline__ void cvt_store4(bf16* dst, float a, float b, float c, float d) {
    v4bf t;
    t[0] = (bf16)a; t[1] = (bf16)b; t[2] = (bf16)c; t[3] = (bf16)d;
    *(v4bf*)dst = t;                       // one ds_store_b64
}

// ---------------------------------------------------------------------------
// Pack a [Kdim, 256] row-major f32 weight into bf16 WMMA B-fragments:
// idx = ((nt*KC + kc)*32 + lane)*16 + e
// N = nt*16 + (lane&15) ; K = kc*32 + (e&7) + 8*(lane>=16) + 16*(e>=8)
// ---------------------------------------------------------------------------
__global__ void pack_w_kernel(const float* __restrict__ W, bf16* __restrict__ P,
                              int Ncols, int KC, int NT) {
    int idx = blockIdx.x * 256 + threadIdx.x;
    int total = NT * KC * 512;
    if (idx >= total) return;
    int e    = idx & 15;
    int lane = (idx >> 4) & 31;
    int kc   = (idx >> 9) % KC;
    int nt   = idx / (KC * 512);
    int N = nt * 16 + (lane & 15);
    int K = kc * 32 + (e & 7) + ((lane & 16) ? 8 : 0) + ((e & 8) ? 16 : 0);
    P[idx] = (bf16)W[(size_t)K * Ncols + N];
}

// ---------------------------------------------------------------------------
// Degree counts: cnt[col[e]] += 1
// ---------------------------------------------------------------------------
__global__ void count_kernel(const int* __restrict__ eidx, float* __restrict__ cnt) {
    int e = blockIdx.x * 256 + threadIdx.x;
    if (e < NE) atomicAdd(&cnt[eidx[NE + e]], 1.0f);
}

// ---------------------------------------------------------------------------
// Edge kernel: 256 edges / block (8 waves x 32 rows), fused
//   h = elu(concat(x[row],ea) @ w1a + b1a) @ w1b + b1b ; sums[col] += h
// ---------------------------------------------------------------------------
__global__ void __launch_bounds__(256)
__attribute__((amdgpu_waves_per_eu(2)))
edge_kernel(const float* __restrict__ x, const int* __restrict__ eidx,
            const float* __restrict__ ea,
            const float* __restrict__ b1a, const float* __restrict__ b1b,
            const bf16* __restrict__ pw1a, const bf16* __restrict__ pw1b,
            float* __restrict__ sums) {
    extern __shared__ __align__(16) char smem[];
    bf16* Ash = (bf16*)smem;              // [256][256] bf16  (128 KB)
    bf16* Hsh = Ash + 256 * 256;          // [256][256] bf16  (128 KB)
    int*  colsh = (int*)(Hsh + 256 * 256);

    const int tid = threadIdx.x;
    const long long e0 = (long long)blockIdx.x * 256;   // NE % 256 == 0

    colsh[tid] = eidx[NE + e0 + tid];

    // gather inputs -> bf16 LDS tile: 1 thread per edge
    {
        const long long ge = e0 + tid;
        const int r = eidx[ge];
        const float* srcx = x + (long long)r * DN;
        const float* srce = ea + ge * DE;
        bf16* dst = Ash + tid * 256;
        #pragma unroll
        for (int k = 0; k < 128; k += 4) {
            float4 v = *(const float4*)(srcx + k);
            cvt_store4(dst + k, v.x, v.y, v.z, v.w);
        }
        #pragma unroll
        for (int k = 0; k < 128; k += 4) {
            float4 v = *(const float4*)(srce + k);
            cvt_store4(dst + 128 + k, v.x, v.y, v.z, v.w);
        }
    }
    __syncthreads();

    const int lane  = tid & 31;
    const int wave  = tid >> 5;
    const int rbase = wave * 32;          // wave's 32 edge rows
    const int nl    = lane & 15;          // M for A-frags, N for C/D frags
    const int khi   = (lane & 16) ? 8 : 0;

    v16bf af0[8], af1[8];

    // ---- GEMM1: [32,256] @ w1a -> elu -> Hsh -----------------------------
    {
        const bf16* ar0 = Ash + (rbase + nl) * 256;
        const bf16* ar1 = ar0 + 16 * 256;
        #pragma unroll
        for (int kc = 0; kc < 8; ++kc) {
            af0[kc] = load_a_frag(ar0, kc * 32 + khi);
            af1[kc] = load_a_frag(ar1, kc * 32 + khi);
        }
        #pragma unroll
        for (int nt = 0; nt < 16; nt += 2) {
            v8f a0 = ZERO8, a1 = ZERO8, a2 = ZERO8, a3 = ZERO8;
            #pragma unroll
            for (int kc = 0; kc < 8; ++kc) {
                v16bf bA = *(const v16bf*)(pw1a + ((((nt    ) * 8 + kc) * 32 + lane) << 4));
                v16bf bB = *(const v16bf*)(pw1a + ((((nt + 1) * 8 + kc) * 32 + lane) << 4));
                a0 = WMMA_BF16(af0[kc], bA, a0);
                a1 = WMMA_BF16(af1[kc], bA, a1);
                a2 = WMMA_BF16(af0[kc], bB, a2);
                a3 = WMMA_BF16(af1[kc], bB, a3);
            }
            const int N0 = nt * 16 + nl;
            const float biasA = b1a[N0];
            const float biasB = b1a[N0 + 16];
            #pragma unroll
            for (int v = 0; v < 8; ++v) {
                const int r0 = (rbase + v + khi) * 256;
                const int r1 = (rbase + 16 + v + khi) * 256;
                Hsh[r0 + N0]      = (bf16)elu1(a0[v] + biasA);
                Hsh[r1 + N0]      = (bf16)elu1(a1[v] + biasA);
                Hsh[r0 + N0 + 16] = (bf16)elu1(a2[v] + biasB);
                Hsh[r1 + N0 + 16] = (bf16)elu1(a3[v] + biasB);
            }
        }
    }
    __syncthreads();

    // ---- GEMM2: h1 @ w1b + b1b, then atomic scatter-add ------------------
    {
        const bf16* hr0 = Hsh + (rbase + nl) * 256;
        const bf16* hr1 = hr0 + 16 * 256;
        #pragma unroll
        for (int kc = 0; kc < 8; ++kc) {
            af0[kc] = load_a_frag(hr0, kc * 32 + khi);
            af1[kc] = load_a_frag(hr1, kc * 32 + khi);
        }
        #pragma unroll
        for (int nt = 0; nt < 16; nt += 2) {
            v8f a0 = ZERO8, a1 = ZERO8, a2 = ZERO8, a3 = ZERO8;
            #pragma unroll
            for (int kc = 0; kc < 8; ++kc) {
                v16bf bA = *(const v16bf*)(pw1b + ((((nt    ) * 8 + kc) * 32 + lane) << 4));
                v16bf bB = *(const v16bf*)(pw1b + ((((nt + 1) * 8 + kc) * 32 + lane) << 4));
                a0 = WMMA_BF16(af0[kc], bA, a0);
                a1 = WMMA_BF16(af1[kc], bA, a1);
                a2 = WMMA_BF16(af0[kc], bB, a2);
                a3 = WMMA_BF16(af1[kc], bB, a3);
            }
            const int N0 = nt * 16 + nl;
            const float biasA = b1b[N0];
            const float biasB = b1b[N0 + 16];
            #pragma unroll
            for (int v = 0; v < 8; ++v) {
                const int c0 = colsh[rbase + v + khi];
                const int c1 = colsh[rbase + 16 + v + khi];
                atomicAdd(sums + (long long)c0 * DH + N0,      a0[v] + biasA);
                atomicAdd(sums + (long long)c1 * DH + N0,      a1[v] + biasA);
                atomicAdd(sums + (long long)c0 * DH + N0 + 16, a2[v] + biasB);
                atomicAdd(sums + (long long)c1 * DH + N0 + 16, a3[v] + biasB);
            }
        }
    }
}

// ---------------------------------------------------------------------------
// Node kernel: 128 nodes / block (4 waves x 32 rows)
//   out = elu(concat(x, sums/max(cnt,1)) @ w2a + b2a) @ w2b + b2b
// ---------------------------------------------------------------------------
__global__ void __launch_bounds__(128)
node_kernel(const float* __restrict__ x, const float* __restrict__ sums,
            const float* __restrict__ cnt,
            const float* __restrict__ b2a, const float* __restrict__ b2b,
            const bf16* __restrict__ pw2a, const bf16* __restrict__ pw2b,
            float* __restrict__ out) {
    extern __shared__ __align__(16) char smem[];
    bf16* Ash = (bf16*)smem;              // [128][384] bf16  (96 KB)
    bf16* Hsh = Ash + 128 * 384;          // [128][256] bf16  (64 KB)

    const int tid = threadIdx.x;
    const int n0 = blockIdx.x * 128;

    // build A tile (clamp OOR nodes; their outputs are never stored)
    {
        int node = n0 + tid; if (node >= NN) node = NN - 1;
        const float* srcx = x + (size_t)node * DN;
        const float* srcs = sums + (size_t)node * DH;
        const float inv = 1.0f / fmaxf(cnt[node], 1.0f);
        bf16* dst = Ash + tid * 384;
        #pragma unroll
        for (int k = 0; k < 128; k += 4) {
            float4 v = *(const float4*)(srcx + k);
            cvt_store4(dst + k, v.x, v.y, v.z, v.w);
        }
        #pragma unroll
        for (int k = 0; k < 256; k += 4) {
            float4 v = *(const float4*)(srcs + k);
            cvt_store4(dst + 128 + k, v.x * inv, v.y * inv, v.z * inv, v.w * inv);
        }
    }
    __syncthreads();

    const int lane  = tid & 31;
    const int wave  = tid >> 5;
    const int rbase = wave * 32;
    const int nl    = lane & 15;
    const int khi   = (lane & 16) ? 8 : 0;

    // ---- GEMM1: [32,384] @ w2a (12 k-chunks) -> elu -> Hsh ---------------
    {
        v16bf af0[12], af1[12];
        const bf16* ar0 = Ash + (rbase + nl) * 384;
        const bf16* ar1 = ar0 + 16 * 384;
        #pragma unroll
        for (int kc = 0; kc < 12; ++kc) {
            af0[kc] = load_a_frag(ar0, kc * 32 + khi);
            af1[kc] = load_a_frag(ar1, kc * 32 + khi);
        }
        #pragma unroll
        for (int nt = 0; nt < 16; nt += 2) {
            v8f a0 = ZERO8, a1 = ZERO8, a2 = ZERO8, a3 = ZERO8;
            #pragma unroll
            for (int kc = 0; kc < 12; ++kc) {
                v16bf bA = *(const v16bf*)(pw2a + ((((nt    ) * 12 + kc) * 32 + lane) << 4));
                v16bf bB = *(const v16bf*)(pw2a + ((((nt + 1) * 12 + kc) * 32 + lane) << 4));
                a0 = WMMA_BF16(af0[kc], bA, a0);
                a1 = WMMA_BF16(af1[kc], bA, a1);
                a2 = WMMA_BF16(af0[kc], bB, a2);
                a3 = WMMA_BF16(af1[kc], bB, a3);
            }
            const int N0 = nt * 16 + nl;
            const float biasA = b2a[N0];
            const float biasB = b2a[N0 + 16];
            #pragma unroll
            for (int v = 0; v < 8; ++v) {
                const int r0 = (rbase + v + khi) * 256;
                const int r1 = (rbase + 16 + v + khi) * 256;
                Hsh[r0 + N0]      = (bf16)elu1(a0[v] + biasA);
                Hsh[r1 + N0]      = (bf16)elu1(a1[v] + biasA);
                Hsh[r0 + N0 + 16] = (bf16)elu1(a2[v] + biasB);
                Hsh[r1 + N0 + 16] = (bf16)elu1(a3[v] + biasB);
            }
        }
    }
    __syncthreads();

    // ---- GEMM2: h @ w2b + b2b -> out -------------------------------------
    {
        v16bf af0[8], af1[8];
        const bf16* hr0 = Hsh + (rbase + nl) * 256;
        const bf16* hr1 = hr0 + 16 * 256;
        #pragma unroll
        for (int kc = 0; kc < 8; ++kc) {
            af0[kc] = load_a_frag(hr0, kc * 32 + khi);
            af1[kc] = load_a_frag(hr1, kc * 32 + khi);
        }
        #pragma unroll
        for (int nt = 0; nt < 16; nt += 2) {
            v8f a0 = ZERO8, a1 = ZERO8, a2 = ZERO8, a3 = ZERO8;
            #pragma unroll
            for (int kc = 0; kc < 8; ++kc) {
                v16bf bA = *(const v16bf*)(pw2b + ((((nt    ) * 8 + kc) * 32 + lane) << 4));
                v16bf bB = *(const v16bf*)(pw2b + ((((nt + 1) * 8 + kc) * 32 + lane) << 4));
                a0 = WMMA_BF16(af0[kc], bA, a0);
                a1 = WMMA_BF16(af1[kc], bA, a1);
                a2 = WMMA_BF16(af0[kc], bB, a2);
                a3 = WMMA_BF16(af1[kc], bB, a3);
            }
            const int N0 = nt * 16 + nl;
            const float biasA = b2b[N0];
            const float biasB = b2b[N0 + 16];
            #pragma unroll
            for (int v = 0; v < 8; ++v) {
                const int node0 = n0 + rbase + v + khi;
                const int node1 = node0 + 16;
                if (node0 < NN) {
                    out[(size_t)node0 * DO + N0]      = a0[v] + biasA;
                    out[(size_t)node0 * DO + N0 + 16] = a2[v] + biasB;
                }
                if (node1 < NN) {
                    out[(size_t)node1 * DO + N0]      = a1[v] + biasA;
                    out[(size_t)node1 * DO + N0 + 16] = a3[v] + biasB;
                }
            }
        }
    }
}

// ---------------------------------------------------------------------------
extern "C" void kernel_launch(void* const* d_in, const int* in_sizes, int n_in,
                              void* d_out, int out_size, void* d_ws, size_t ws_size,
                              hipStream_t stream) {
    const float* x    = (const float*)d_in[0];
    const int*   eidx = (const int*)  d_in[1];   // [2, NE] flattened
    const float* ea   = (const float*)d_in[2];
    const float* w1a  = (const float*)d_in[5];
    const float* b1a  = (const float*)d_in[6];
    const float* w1b  = (const float*)d_in[7];
    const float* b1b  = (const float*)d_in[8];
    const float* w2a  = (const float*)d_in[9];
    const float* b2a  = (const float*)d_in[10];
    const float* w2b  = (const float*)d_in[11];
    const float* b2b  = (const float*)d_in[12];
    float* out = (float*)d_out;

    // workspace layout
    char* ws = (char*)d_ws;
    float* sums = (float*)ws;                                   // [NN, 256] f32
    float* cnt  = (float*)(ws + (size_t)NN * DH * 4);           // [NN] f32
    size_t o = (((size_t)(NN * DH + NN) * 4) + 255) & ~(size_t)255;
    bf16* pw1a = (bf16*)(ws + o); o += (size_t)16 * 8  * 512 * 2;
    bf16* pw1b = (bf16*)(ws + o); o += (size_t)16 * 8  * 512 * 2;
    bf16* pw2a = (bf16*)(ws + o); o += (size_t)16 * 12 * 512 * 2;
    bf16* pw2b = (bf16*)(ws + o);

    // zero accumulators (sums + cnt are contiguous)
    hipMemsetAsync(sums, 0, (size_t)(NN * DH + NN) * 4, stream);

    // pack weights -> bf16 WMMA fragment order
    pack_w_kernel<<<(16 * 8  * 512 + 255) / 256, 256, 0, stream>>>(w1a, pw1a, 256, 8, 16);
    pack_w_kernel<<<(16 * 8  * 512 + 255) / 256, 256, 0, stream>>>(w1b, pw1b, 256, 8, 16);
    pack_w_kernel<<<(16 * 12 * 512 + 255) / 256, 256, 0, stream>>>(w2a, pw2a, 256, 12, 16);
    pack_w_kernel<<<(16 * 8  * 512 + 255) / 256, 256, 0, stream>>>(w2b, pw2b, 256, 8, 16);

    // degree counts
    count_kernel<<<(NE + 255) / 256, 256, 0, stream>>>(eidx, cnt);

    // fused edge MLP + scatter-add (8 waves x 32 rows, 263168 B dynamic LDS)
    const int edgeLds = 256 * 256 * 2 * 2 + 256 * 4;
    hipFuncSetAttribute((const void*)edge_kernel,
                        hipFuncAttributeMaxDynamicSharedMemorySize, edgeLds);
    edge_kernel<<<NE / 256, 256, edgeLds, stream>>>(x, eidx, ea, b1a, b1b,
                                                    pw1a, pw1b, sums);

    // node MLP (4 waves x 32 rows, 163840 B dynamic LDS)
    const int nodeLds = 128 * 384 * 2 + 128 * 256 * 2;
    hipFuncSetAttribute((const void*)node_kernel,
                        hipFuncAttributeMaxDynamicSharedMemorySize, nodeLds);
    node_kernel<<<(NN + 127) / 128, 128, nodeLds, stream>>>(x, sums, cnt, b2a, b2b,
                                                            pw2a, pw2b, out);
}
